// DeepSeekBlock_21294447853773
// MI455X (gfx1250) — matrix operand order repaired and checked by reading, not compiled, not run on hardware
//
#include <hip/hip_runtime.h>
#include <hip/hip_bf16.h>
#include <math.h>

#define BB 2
#define TT 2048
#define HD 1024
#define LD 256
#define FD 2048
#define EE 7
#define KC 64   // K-chunk staged in LDS per iteration (64 * 2B = 128B per row)

typedef __attribute__((ext_vector_type(16))) __bf16 v16bf;
typedef __attribute__((ext_vector_type(8)))  float  v8f;
typedef __attribute__((ext_vector_type(4)))  unsigned int u32x4;
typedef __hip_bfloat16 bf16s;

union FragU { u32x4 u[2]; v16bf v; };

// ---------------------------------------------------------------------------
// WMMA fragment loaders (bf16, K contiguous).
// A (from LDS tile, row stride KC): lane m=l&15, kh=l>>4;
//    elems 0..15 -> K = 16*kh+0..15.
// B (global, lane n holds contiguous row of W, i.e. B = W^T):
//    elems 0..15 -> K = 16*(l>>4)+0..15.
// ---------------------------------------------------------------------------
__device__ inline v16bf load_a_lds(const bf16s* At, int row, int koff, int lane) {
  const int kh = lane >> 4;
  const bf16s* p = At + row * KC + koff + 16 * kh;
  FragU f;
  f.u[0] = *(const u32x4*)(p);
  f.u[1] = *(const u32x4*)(p + 8);
  return f.v;
}

__device__ inline v16bf load_b_frag(const bf16s* Bw, int ldb, int n0, int k0, int lane) {
  const int n = lane & 15, kb = (lane >> 4) * 16;
  const bf16s* p = Bw + (size_t)(n0 + n) * ldb + (k0 + kb);
  FragU f;
  f.u[0] = *(const u32x4*)(p);
  f.u[1] = *(const u32x4*)(p + 8);
  return f.v;
}

// Epilogue modes
// 0: store bf16            1: store bf16 transposed    2: causal scores fp32
// 3: fp32 store + addsrc   4: bf16 silu(acc)*aux       5: fp32 += rowscale*acc
// 6: plain fp32 store
template<int EPI>
__global__ __launch_bounds__(256) void gemm_bt(
    const bf16s* __restrict__ A, const bf16s* __restrict__ Bw,
    void* __restrict__ Cv, const float* __restrict__ aux, int auxstride,
    int M, int N, int Kd, int lda, int ldb, int ldc,
    long long sA, long long sB, long long sC, float scale)
{
  // double-buffered A tile: 2 x (64 rows x KC bf16) = 2 x 8KB
  __shared__ bf16s shA[2][64 * KC];

  const int tid  = threadIdx.x;
  const int lane = tid & 31;
  const int wv   = tid >> 5;
  const int n0   = (blockIdx.x * 8 + wv) * 16;   // 16-wide N tile per wave
  const int m0   = blockIdx.y * 64;              // 64-tall M macro-tile per block
  const int bz   = blockIdx.z;
  // whole block above the causal diagonal: exit before any barrier/async op
  if (EPI == 2 && (int)blockIdx.x * 128 > m0 + 63) return;
  // wave-level mask (wave-uniform; masked waves still stage + barrier)
  const bool active = !(EPI == 2 && n0 > m0 + 63);

  const bf16s* Ab  = A  + (size_t)bz * sA;
  const bf16s* Bwb = Bw + (size_t)bz * sB;

  // low 32 bits of a generic pointer to LDS == LDS byte offset (ISA 10.2)
  const unsigned ldsbase = (unsigned)(size_t)(&shA[0][0]);

  // async staging of one 64 x KC A-tile: 256 threads x 2 x 16B = 8KB
  auto stage = [&](int c, int buf) {
    #pragma unroll
    for (int j = 0; j < 2; ++j) {
      const int o   = tid * 16 + j * 4096;   // byte offset inside tile
      const int row = o >> 7;                // KC*2 = 128 bytes per row
      const int kb  = o & 127;
      const unsigned long long ga =
          (unsigned long long)(const char*)(Ab + (size_t)(m0 + row) * lda) +
          (unsigned long long)(c * (KC * 2) + kb);
      const unsigned lo = ldsbase + (unsigned)(buf * 8192 + o);
      asm volatile("global_load_async_to_lds_b128 %0, %1, off"
                   :: "v"(lo), "v"(ga) : "memory");
    }
  };

  v8f zero = {};
  v8f acc[4];
  #pragma unroll
  for (int i = 0; i < 4; ++i) acc[i] = zero;

  const int nchunks = Kd / KC;
  stage(0, 0);
  for (int c = 0; c < nchunks; ++c) {
    asm volatile("s_wait_asynccnt 0x0" ::: "memory");  // my async writes done
    __syncthreads();                                   // everyone's writes visible
    if (c + 1 < nchunks) stage(c + 1, (c + 1) & 1);    // prefetch next chunk
    if (active) {
      const bf16s* At = &shA[c & 1][0];
      #pragma unroll
      for (int ks = 0; ks < KC / 32; ++ks) {
        v16bf bfrag = load_b_frag(Bwb, ldb, n0, c * KC + ks * 32, lane);
        #pragma unroll
        for (int mt = 0; mt < 4; ++mt) {
          v16bf afrag = load_a_lds(At, mt * 16 + (lane & 15), ks * 32, lane);
          acc[mt] = __builtin_amdgcn_wmma_f32_16x16x32_bf16(
              false, afrag, false, bfrag, (short)0, acc[mt], false, false);
        }
      }
    }
    __syncthreads();  // safe to overwrite buf (c&1) two iterations later
  }

  if (!active) return;

  const int cn = n0 + (lane & 15);
  const int mr = (lane >> 4) * 8;
  #pragma unroll
  for (int mt = 0; mt < 4; ++mt) {
    #pragma unroll
    for (int r = 0; r < 8; ++r) {
      const int cm = m0 + mt * 16 + mr + r;
      const float v = acc[mt][r] * scale;
      if (EPI == 0) {
        bf16s* C = (bf16s*)Cv + (size_t)bz * sC;
        C[(size_t)cm * ldc + cn] = __float2bfloat16(v);
      } else if (EPI == 1) {
        bf16s* C = (bf16s*)Cv + (size_t)bz * sC;
        C[(size_t)cn * ldc + cm] = __float2bfloat16(v);
      } else if (EPI == 2) {
        float* C = (float*)Cv + (size_t)bz * sC;
        if (cn <= cm) C[(size_t)cm * ldc + cn] = v;       // masked cols never read
      } else if (EPI == 3) {
        float* C = (float*)Cv;
        const size_t idx = (size_t)cm * ldc + cn;
        C[idx] = aux[idx] + v;
      } else if (EPI == 4) {
        bf16s* C = (bf16s*)Cv;
        const size_t idx = (size_t)cm * ldc + cn;
        const float sg = v / (1.f + __expf(-v));          // silu(gate)
        C[idx] = __float2bfloat16(sg * aux[idx]);
      } else if (EPI == 5) {
        float* C = (float*)Cv;
        const float rs = aux ? aux[(size_t)cm * auxstride] : 1.f;
        C[(size_t)cm * ldc + cn] += rs * v;
      } else {
        float* C = (float*)Cv;
        C[(size_t)cm * ldc + cn] = v;
      }
    }
  }
}

// ---------------------------------------------------------------------------
// Elementwise / reduction kernels
// ---------------------------------------------------------------------------
__global__ __launch_bounds__(256) void cvt_bf16_k(const float* __restrict__ s,
                                                  bf16s* __restrict__ d, int n) {
  const int i = blockIdx.x * 256 + threadIdx.x;
  if (i < n) d[i] = __float2bfloat16(s[i]);
}

__global__ __launch_bounds__(256) void ln_bf16_k(const float* __restrict__ x,
                                                 const float* __restrict__ w,
                                                 bf16s* __restrict__ o, int hd) {
  __shared__ float s1[256], s2[256];
  const int row = blockIdx.x, tid = threadIdx.x;
  const float* xr = x + (size_t)row * hd;
  float a = 0.f, b = 0.f;
  for (int i = tid; i < hd; i += 256) { float v = xr[i]; a += v; b += v * v; }
  s1[tid] = a; s2[tid] = b; __syncthreads();
  for (int o2 = 128; o2 > 0; o2 >>= 1) {
    if (tid < o2) { s1[tid] += s1[tid + o2]; s2[tid] += s2[tid + o2]; }
    __syncthreads();
  }
  const float mean = s1[0] / hd;
  const float var  = s2[0] / hd - mean * mean;
  const float inv  = rsqrtf(var + 1e-5f);
  bf16s* orow = o + (size_t)row * hd;
  for (int i = tid; i < hd; i += 256)
    orow[i] = __float2bfloat16((xr[i] - mean) * inv * w[i]);
}

__global__ __launch_bounds__(256) void rope_qk_k(const bf16s* __restrict__ qr,
                                                 const bf16s* __restrict__ kr,
                                                 bf16s* __restrict__ q,
                                                 bf16s* __restrict__ k,
                                                 int Tlen, int hd) {
  const int row = blockIdx.x;
  const int t = row % Tlen;
  const int half = hd >> 1;
  const size_t base = (size_t)row * hd;
  for (int i = threadIdx.x; i < hd; i += 256) {
    const int fi = (i < half) ? i : i - half;
    const float invf = __expf(-((float)(2 * fi) / hd) * 9.2103403719761836f); // ln(1e4)
    const float ang = (float)t * invf;
    const float c = cosf(ang), s = sinf(ang);
    const float qv = __bfloat162float(qr[base + i]);
    const float kv = __bfloat162float(kr[base + i]);
    const float qrt = (i < half) ? -__bfloat162float(qr[base + i + half])
                                 :  __bfloat162float(qr[base + i - half]);
    const float krt = (i < half) ? -__bfloat162float(kr[base + i + half])
                                 :  __bfloat162float(kr[base + i - half]);
    q[base + i] = __float2bfloat16(qv * c + qrt * s);
    k[base + i] = __float2bfloat16(kv * c + krt * s);
  }
}

__global__ __launch_bounds__(256) void softmax_causal_k(const float* __restrict__ sc,
                                                        bf16s* __restrict__ p, int Tlen) {
  __shared__ float red[256];
  const int row = blockIdx.x;                 // b*T + m
  const int b = row / Tlen, m = row % Tlen, tid = threadIdx.x;
  const float* sr = sc + (size_t)b * Tlen * Tlen + (size_t)m * Tlen;
  bf16s* pr = p + (size_t)b * Tlen * Tlen + (size_t)m * Tlen;
  float mx = -3.0e38f;
  for (int j = tid; j <= m; j += 256) mx = fmaxf(mx, sr[j]);
  red[tid] = mx; __syncthreads();
  for (int o = 128; o > 0; o >>= 1) {
    if (tid < o) red[tid] = fmaxf(red[tid], red[tid + o]);
    __syncthreads();
  }
  mx = red[0]; __syncthreads();
  float sum = 0.f;
  for (int j = tid; j <= m; j += 256) sum += __expf(sr[j] - mx);
  red[tid] = sum; __syncthreads();
  for (int o = 128; o > 0; o >>= 1) {
    if (tid < o) red[tid] += red[tid + o];
    __syncthreads();
  }
  const float rinv = 1.f / red[0];
  for (int j = tid; j < Tlen; j += 256)
    pr[j] = __float2bfloat16(j <= m ? __expf(sr[j] - mx) * rinv : 0.f);
}

__global__ __launch_bounds__(256) void router_logits_k(const bf16s* __restrict__ h,
                                                       const float* __restrict__ w,
                                                       const float* __restrict__ bias,
                                                       float* __restrict__ probs,
                                                       int ntok, int hd) {
  const int idx = blockIdx.x * 256 + threadIdx.x;
  if (idx >= ntok * EE) return;
  const int tok = idx / EE, e = idx % EE;
  const bf16s* hr = h + (size_t)tok * hd;
  const float* wr = w + (size_t)e * hd;
  float a = 0.f;
  for (int i = 0; i < hd; ++i) a += __bfloat162float(hr[i]) * wr[i];
  a += bias[e];
  probs[idx] = 1.f / (1.f + __expf(-a));
}

__global__ __launch_bounds__(256) void topk_combine_k(const float* __restrict__ probs,
                                                      float* __restrict__ comb, int ntok) {
  const int tok = blockIdx.x * 256 + threadIdx.x;
  if (tok >= ntok) return;
  const float* p = probs + (size_t)tok * EE;
  float* c = comb + (size_t)tok * EE;
  int i1 = 0; float v1 = -1.f;
  for (int e = 0; e < EE; ++e) { float v = p[e]; if (v > v1) { v1 = v; i1 = e; } }
  int i2 = 0; float v2 = -1.f;
  for (int e = 0; e < EE; ++e) { if (e == i1) continue; float v = p[e]; if (v > v2) { v2 = v; i2 = e; } }
  for (int e = 0; e < EE; ++e) c[e] = 0.f;
  c[i1] = v1; c[i2] = v2;
}

// ---------------------------------------------------------------------------
// Host-side dispatch
// ---------------------------------------------------------------------------
static void gemm_launch(int epi, const bf16s* A, const bf16s* Bw, void* C,
                        const float* aux, int auxstride,
                        int M, int N, int Kd, int lda, int ldb, int ldc,
                        long long sA, long long sB, long long sC,
                        float scale, int batch, hipStream_t st) {
  dim3 g(N / 128, M / 64, batch), b(256, 1, 1);
  switch (epi) {
    case 0: gemm_bt<0><<<g, b, 0, st>>>(A, Bw, C, aux, auxstride, M, N, Kd, lda, ldb, ldc, sA, sB, sC, scale); break;
    case 1: gemm_bt<1><<<g, b, 0, st>>>(A, Bw, C, aux, auxstride, M, N, Kd, lda, ldb, ldc, sA, sB, sC, scale); break;
    case 2: gemm_bt<2><<<g, b, 0, st>>>(A, Bw, C, aux, auxstride, M, N, Kd, lda, ldb, ldc, sA, sB, sC, scale); break;
    case 3: gemm_bt<3><<<g, b, 0, st>>>(A, Bw, C, aux, auxstride, M, N, Kd, lda, ldb, ldc, sA, sB, sC, scale); break;
    case 4: gemm_bt<4><<<g, b, 0, st>>>(A, Bw, C, aux, auxstride, M, N, Kd, lda, ldb, ldc, sA, sB, sC, scale); break;
    case 5: gemm_bt<5><<<g, b, 0, st>>>(A, Bw, C, aux, auxstride, M, N, Kd, lda, ldb, ldc, sA, sB, sC, scale); break;
    default: gemm_bt<6><<<g, b, 0, st>>>(A, Bw, C, aux, auxstride, M, N, Kd, lda, ldb, ldc, sA, sB, sC, scale); break;
  }
}

extern "C" void kernel_launch(void* const* d_in, const int* in_sizes, int n_in,
                              void* d_out, int out_size, void* d_ws, size_t ws_size,
                              hipStream_t stream) {
  (void)in_sizes; (void)n_in; (void)out_size; (void)ws_size;
  const float* x        = (const float*)d_in[0];
  const float* ln1_w    = (const float*)d_in[1];
  const float* ln2_w    = (const float*)d_in[2];
  const float* kv_d     = (const float*)d_in[3];
  const float* q_d      = (const float*)d_in[4];
  // d_in[5] k_proj_u, d_in[6] q_proj_u: dead in the reference (overwritten by rope path)
  const float* v_u      = (const float*)d_in[7];
  const float* rope_kw  = (const float*)d_in[8];
  const float* rope_qw  = (const float*)d_in[9];
  const float* o_proj   = (const float*)d_in[10];
  const float* router_w = (const float*)d_in[11];
  const float* r_bias   = (const float*)d_in[12];
  const float* sh_gate  = (const float*)d_in[13];
  const float* sh_up    = (const float*)d_in[14];
  const float* sh_down  = (const float*)d_in[15];
  const float* ex_gate  = (const float*)d_in[16];
  const float* ex_up    = (const float*)d_in[17];
  const float* ex_down  = (const float*)d_in[18];
  float* out = (float*)d_out;

  const int BT  = BB * TT;
  const size_t LH  = (size_t)LD * HD;
  const size_t HH2 = (size_t)HD * HD;
  const size_t FH  = (size_t)FD * HD;
  const size_t EFH = (size_t)EE * FH;
  const size_t BTH = (size_t)BT * HD;
  const size_t BTL = (size_t)BT * LD;
  const size_t BTT = (size_t)BB * TT * TT;
  const size_t BTE = (size_t)BT * EE;

  char* base = (char*)d_ws;
  size_t off = 0;
  auto alloc = [&](size_t bytes) -> char* {
    char* p = base + off;
    off = (off + bytes + 255) & ~(size_t)255;
    return p;
  };

  // bf16 weight copies (fit in 192MB L2 for heavy reuse)
  bf16s* wb_kv = (bf16s*)alloc(LH * 2);
  bf16s* wb_q  = (bf16s*)alloc(LH * 2);
  bf16s* wb_v  = (bf16s*)alloc(LH * 2);
  bf16s* wb_rq = (bf16s*)alloc(LH * 2);
  bf16s* wb_rk = (bf16s*)alloc(HH2 * 2);
  bf16s* wb_o  = (bf16s*)alloc(HH2 * 2);
  bf16s* wb_sg = (bf16s*)alloc(FH * 2);
  bf16s* wb_su = (bf16s*)alloc(FH * 2);
  bf16s* wb_sd = (bf16s*)alloc(FH * 2);
  bf16s* wb_eg = (bf16s*)alloc(EFH * 2);
  bf16s* wb_eu = (bf16s*)alloc(EFH * 2);
  bf16s* wb_ed = (bf16s*)alloc(EFH * 2);
  // activations
  bf16s* h1b    = (bf16s*)alloc(BTH * 2);
  bf16s* kvlatb = (bf16s*)alloc(BTL * 2);
  bf16s* qlatb  = (bf16s*)alloc(BTL * 2);
  bf16s* vTb    = (bf16s*)alloc(BTH * 2);   // [B][H][T] transposed
  bf16s* qrb    = (bf16s*)alloc(BTH * 2);
  bf16s* krb    = (bf16s*)alloc(BTH * 2);
  bf16s* qb     = (bf16s*)alloc(BTH * 2);
  bf16s* kb     = (bf16s*)alloc(BTH * 2);
  float* bigF   = (float*)alloc(BTT * 4);   // scores, later reused as MoE "up" (same size)
  bf16s* bigB   = (bf16s*)alloc(BTT * 2);   // attn probs, later reused as MoE act
  bf16s* yb     = (bf16s*)alloc(BTH * 2);
  bf16s* h2b    = (bf16s*)alloc(BTH * 2);
  float* probsf = (float*)alloc(BTE * 4);
  float* combf  = (float*)alloc(BTE * 4);

  auto cvt = [&](const float* s, bf16s* d, size_t n) {
    cvt_bf16_k<<<dim3((unsigned)((n + 255) / 256)), dim3(256), 0, stream>>>(s, d, (int)n);
  };

  // --- weight conversion ---
  cvt(kv_d, wb_kv, LH);     cvt(q_d, wb_q, LH);       cvt(v_u, wb_v, LH);
  cvt(rope_qw, wb_rq, LH);  cvt(rope_kw, wb_rk, HH2); cvt(o_proj, wb_o, HH2);
  cvt(sh_gate, wb_sg, FH);  cvt(sh_up, wb_su, FH);    cvt(sh_down, wb_sd, FH);
  cvt(ex_gate, wb_eg, EFH); cvt(ex_up, wb_eu, EFH);   cvt(ex_down, wb_ed, EFH);

  // --- attention ---
  ln_bf16_k<<<BT, 256, 0, stream>>>(x, ln1_w, h1b, HD);
  gemm_launch(0, h1b, wb_kv, kvlatb, nullptr, 0, BT, LD, HD, HD, HD, LD, 0, 0, 0, 1.f, 1, stream);
  gemm_launch(0, h1b, wb_q,  qlatb,  nullptr, 0, BT, LD, HD, HD, HD, LD, 0, 0, 0, 1.f, 1, stream);
  gemm_launch(1, kvlatb, wb_v, vTb, nullptr, 0, TT, HD, LD, LD, LD, TT,
              (long long)TT * LD, 0, (long long)HD * TT, 1.f, BB, stream);
  gemm_launch(0, qlatb, wb_rq, qrb, nullptr, 0, BT, HD, LD, LD, LD, HD, 0, 0, 0, 1.f, 1, stream);
  gemm_launch(0, h1b,   wb_rk, krb, nullptr, 0, BT, HD, HD, HD, HD, HD, 0, 0, 0, 1.f, 1, stream);
  rope_qk_k<<<BT, 256, 0, stream>>>(qrb, krb, qb, kb, TT, HD);
  gemm_launch(2, qb, kb, bigF, nullptr, 0, TT, TT, HD, HD, HD, TT,
              (long long)TT * HD, (long long)TT * HD, (long long)TT * TT, 0.03125f, BB, stream);
  softmax_causal_k<<<BT, 256, 0, stream>>>(bigF, bigB, TT);
  gemm_launch(0, bigB, vTb, yb, nullptr, 0, TT, HD, TT, TT, TT, HD,
              (long long)TT * TT, (long long)HD * TT, (long long)TT * HD, 1.f, BB, stream);
  gemm_launch(3, yb, wb_o, out, x, 0, BT, HD, HD, HD, HD, HD, 0, 0, 0, 1.f, 1, stream);

  // --- MoE ---
  ln_bf16_k<<<BT, 256, 0, stream>>>(out, ln2_w, h2b, HD);
  router_logits_k<<<(BT * EE + 255) / 256, 256, 0, stream>>>(h2b, router_w, r_bias, probsf, BT, HD);
  topk_combine_k<<<(BT + 255) / 256, 256, 0, stream>>>(probsf, combf, BT);

  // shared expert
  gemm_launch(6, h2b, wb_su, bigF, nullptr, 0, BT, FD, HD, HD, HD, FD, 0, 0, 0, 1.f, 1, stream);
  gemm_launch(4, h2b, wb_sg, bigB, bigF, 0, BT, FD, HD, HD, HD, FD, 0, 0, 0, 1.f, 1, stream);
  gemm_launch(5, bigB, wb_sd, out, nullptr, 0, BT, HD, FD, FD, FD, HD, 0, 0, 0, 1.f, 1, stream);

  // routed experts, weighted by top-2 combine coefficients
  for (int e = 0; e < EE; ++e) {
    const bf16s* eg = wb_eg + (size_t)e * FH;
    const bf16s* eu = wb_eu + (size_t)e * FH;
    const bf16s* ed = wb_ed + (size_t)e * FH;
    gemm_launch(6, h2b, eu, bigF, nullptr, 0, BT, FD, HD, HD, HD, FD, 0, 0, 0, 1.f, 1, stream);
    gemm_launch(4, h2b, eg, bigB, bigF, 0, BT, FD, HD, HD, HD, FD, 0, 0, 0, 1.f, 1, stream);
    gemm_launch(5, bigB, ed, out, combf + e, EE, BT, HD, FD, FD, FD, HD, 0, 0, 0, 1.f, 1, stream);
  }
}
